// CustomLongformerClassifier_1271310319963
// MI455X (gfx1250) — compile-verified, matrix operand order checked by reading, not run on hardware
//
#include <hip/hip_runtime.h>
#include <cstdint>
#include <cstddef>

// ---------------------------------------------------------------------------
// Problem constants (from reference)
// ---------------------------------------------------------------------------
static constexpr int Bb  = 2;
static constexpr int Ss  = 4096;
static constexpr int Dd  = 768;
static constexpr int Hh  = 12;
static constexpr int DHh = 64;
static constexpr int Ll  = 2;
static constexpr int Ff  = 3072;
static constexpr int Cc  = 256;        // attention block size
static constexpr int NBb = Ss / Cc;    // 16
static constexpr int Mm  = Bb * Ss;    // 8192 rows

typedef __bf16 bf16_t;
typedef __bf16 v16bf __attribute__((ext_vector_type(16)));
typedef float  v8f   __attribute__((ext_vector_type(8)));

union Frag { v16bf v; uint4 u[2]; };

__device__ __forceinline__ v8f wmma_bf16(v16bf a, v16bf b, v8f c) {
  // D(f32 16x16) = A(bf16 16x32) * B(bf16 32x16) + C
  return __builtin_amdgcn_wmma_f32_16x16x32_bf16(false, a, false, b,
                                                 (short)0, c, false, false);
}

__device__ __forceinline__ v8f vzero8() {
  v8f z = {0.f, 0.f, 0.f, 0.f, 0.f, 0.f, 0.f, 0.f};
  return z;
}

// CDNA5 async global->LDS copy (ASYNCcnt-tracked; no VGPR round trip).
// ldsaddr: byte offset inside LDS (low 32 bits of a generic __shared__ ptr).
__device__ __forceinline__ void async_copy_b128(uint32_t ldsaddr, uint64_t gaddr) {
  asm volatile("global_load_async_to_lds_b128 %0, %1, off"
               :: "v"(ldsaddr), "v"(gaddr) : "memory");
}
__device__ __forceinline__ void wait_async_le4() {
  asm volatile("s_wait_asynccnt 0x4" ::: "memory");
}
__device__ __forceinline__ void wait_async_0() {
  asm volatile("s_wait_asynccnt 0x0" ::: "memory");
}
__device__ __forceinline__ uint32_t lds_off(const void* p) {
  return (uint32_t)(uintptr_t)p;
}

// ---------------------------------------------------------------------------
// fp32 [K,N] -> bf16 [N,K] transpose (weight prep, one time)
// ---------------------------------------------------------------------------
__global__ void transpose_bf16_kernel(const float* __restrict__ in,
                                      bf16_t* __restrict__ out,
                                      int Kd, int Nd) {
  size_t i = (size_t)blockIdx.x * 256 + threadIdx.x;
  size_t cnt = (size_t)Kd * Nd;
  if (i < cnt) {
    int n  = (int)(i / Kd);
    int kk = (int)(i % Kd);
    out[i] = (bf16_t)in[(size_t)kk * Nd + n];
  }
}

// ---------------------------------------------------------------------------
// Embedding gather + LayerNorm  (one block per row)
// ---------------------------------------------------------------------------
__global__ __launch_bounds__(256)
void embed_ln_kernel(const int* __restrict__ ids,
                     const float* __restrict__ tok,
                     const float* __restrict__ pos,
                     const float* __restrict__ g,
                     const float* __restrict__ b,
                     float* __restrict__ hF, bf16_t* __restrict__ hB) {
  __shared__ float red[256];
  const int row = blockIdx.x;          // 0..Mm-1
  const int s   = row & (Ss - 1);
  const int t   = threadIdx.x;
  const int id  = ids[row];
  float v[3]; float lsum = 0.f;
#pragma unroll
  for (int i = 0; i < 3; ++i) {
    int dd = t + i * 256;
    v[i] = tok[(size_t)id * Dd + dd] + pos[(size_t)s * Dd + dd];
    lsum += v[i];
  }
  red[t] = lsum; __syncthreads();
  for (int o = 128; o > 0; o >>= 1) { if (t < o) red[t] += red[t + o]; __syncthreads(); }
  const float mean = red[0] / (float)Dd;
  __syncthreads();
  float lv = 0.f;
#pragma unroll
  for (int i = 0; i < 3; ++i) { float c = v[i] - mean; lv += c * c; }
  red[t] = lv; __syncthreads();
  for (int o = 128; o > 0; o >>= 1) { if (t < o) red[t] += red[t + o]; __syncthreads(); }
  const float rstd = rsqrtf(red[0] / (float)Dd + 1e-5f);
#pragma unroll
  for (int i = 0; i < 3; ++i) {
    int dd = t + i * 256;
    float o = (v[i] - mean) * rstd * g[dd] + b[dd];
    hF[(size_t)row * Dd + dd] = o;
    hB[(size_t)row * Dd + dd] = (bf16_t)o;
  }
}

// ---------------------------------------------------------------------------
// h = LN(x + y)   (one block per row); writes fp32 + bf16
// ---------------------------------------------------------------------------
__global__ __launch_bounds__(256)
void add_ln_kernel(const float* __restrict__ x, const float* __restrict__ y,
                   const float* __restrict__ g, const float* __restrict__ b,
                   float* __restrict__ hF, bf16_t* __restrict__ hB) {
  __shared__ float red[256];
  const int row = blockIdx.x;
  const int t   = threadIdx.x;
  float v[3]; float lsum = 0.f;
#pragma unroll
  for (int i = 0; i < 3; ++i) {
    int dd = t + i * 256;
    v[i] = x[(size_t)row * Dd + dd] + y[(size_t)row * Dd + dd];
    lsum += v[i];
  }
  red[t] = lsum; __syncthreads();
  for (int o = 128; o > 0; o >>= 1) { if (t < o) red[t] += red[t + o]; __syncthreads(); }
  const float mean = red[0] / (float)Dd;
  __syncthreads();
  float lv = 0.f;
#pragma unroll
  for (int i = 0; i < 3; ++i) { float c = v[i] - mean; lv += c * c; }
  red[t] = lv; __syncthreads();
  for (int o = 128; o > 0; o >>= 1) { if (t < o) red[t] += red[t + o]; __syncthreads(); }
  const float rstd = rsqrtf(red[0] / (float)Dd + 1e-5f);
#pragma unroll
  for (int i = 0; i < 3; ++i) {
    int dd = t + i * 256;
    float o = (v[i] - mean) * rstd * g[dd] + b[dd];
    hF[(size_t)row * Dd + dd] = o;
    hB[(size_t)row * Dd + dd] = (bf16_t)o;
  }
}

// ---------------------------------------------------------------------------
// bf16 WMMA GEMM:  out[M,N] = A[M,K] @ Wt[N,K]^T + bias  (+scale, +GELU)
// BM=128 BN=128 BK=32, 256 threads = 8 waves, each wave -> 32x64 output.
// Double-buffered async global->LDS pipeline (ASYNCcnt), last tile peeled
// so the steady-state loop is branchless.
// ---------------------------------------------------------------------------
#define GEMM_LDA 40   // bf16 row stride (32 + 8 pad) -> 80B, 16B aligned
__global__ __launch_bounds__(256)
void gemm_bf16_kernel(const bf16_t* __restrict__ A,
                      const bf16_t* __restrict__ Wt,
                      const float*  __restrict__ bias,
                      float*  __restrict__ outF,
                      bf16_t* __restrict__ outB,
                      int M, int N, int K, float scale, int act) {
  __shared__ bf16_t As[2][128 * GEMM_LDA];
  __shared__ bf16_t Bs[2][128 * GEMM_LDA];

  const int t    = threadIdx.x;
  const int lane = t & 31;
  const int wave = t >> 5;
  const int wm   = wave & 3;   // 4 row groups of 32
  const int wn   = wave >> 2;  // 2 col groups of 64
  const int m0   = blockIdx.x * 128;
  const int n0   = blockIdx.y * 128;
  const int fr   = lane & 15;
  const int fh   = lane >> 4;

  const int lrow = t >> 2;          // 0..63 (rows lrow and lrow+64)
  const int lcol = (t & 3) * 8;     // bf16 offset inside 32-wide K tile

  // issue one K-tile (4 async b128 copies per thread: 2 A rows + 2 B rows)
  auto issue_tile = [&](int buf, int k0) {
    const bf16_t* gA = A  + (size_t)(m0 + lrow) * K + k0 + lcol;
    const bf16_t* gB = Wt + (size_t)(n0 + lrow) * K + k0 + lcol;
    uint32_t la = lds_off(&As[buf][lrow * GEMM_LDA + lcol]);
    uint32_t lb = lds_off(&Bs[buf][lrow * GEMM_LDA + lcol]);
    const uint32_t lstep = 64 * GEMM_LDA * (uint32_t)sizeof(bf16_t);
    async_copy_b128(la,         (uint64_t)(uintptr_t)gA);
    async_copy_b128(la + lstep, (uint64_t)(uintptr_t)(gA + (size_t)64 * K));
    async_copy_b128(lb,         (uint64_t)(uintptr_t)gB);
    async_copy_b128(lb + lstep, (uint64_t)(uintptr_t)(gB + (size_t)64 * K));
  };

  v8f acc[2][4];
#pragma unroll
  for (int mi = 0; mi < 2; ++mi)
#pragma unroll
    for (int ni = 0; ni < 4; ++ni) acc[mi][ni] = vzero8();

  auto compute_tile = [&](int buf) {
    Frag a[2], b[4];
#pragma unroll
    for (int mi = 0; mi < 2; ++mi) {
      const bf16_t* p = &As[buf][(wm * 32 + mi * 16 + fr) * GEMM_LDA];
      a[mi].u[0] = *(const uint4*)(p + fh * 8);
      a[mi].u[1] = *(const uint4*)(p + 16 + fh * 8);
    }
#pragma unroll
    for (int ni = 0; ni < 4; ++ni) {
      const bf16_t* p = &Bs[buf][(wn * 64 + ni * 16 + fr) * GEMM_LDA];
      b[ni].u[0] = *(const uint4*)(p + fh * 8);
      b[ni].u[1] = *(const uint4*)(p + 16 + fh * 8);
    }
#pragma unroll
    for (int mi = 0; mi < 2; ++mi)
#pragma unroll
      for (int ni = 0; ni < 4; ++ni)
        acc[mi][ni] = wmma_bf16(a[mi].v, b[ni].v, acc[mi][ni]);
  };

  const int NT = K / 32;
  issue_tile(0, 0);

  // steady state: branchless issue-next / wait-current / compute-current
  for (int it = 0; it < NT - 1; ++it) {
    const int buf = it & 1;
    __syncthreads();                       // buf^1 free (last read in previous iter)
    issue_tile(buf ^ 1, (it + 1) * 32);    // prefetch next tile
    wait_async_le4();                      // oldest 4 ops (tile `it`) complete
    __syncthreads();                       // tile `it` visible to all waves
    compute_tile(buf);
  }
  // peeled final tile
  __syncthreads();
  wait_async_0();
  __syncthreads();
  compute_tile((NT - 1) & 1);

  // epilogue: bias -> scale -> (gelu) -> store fp32/bf16
#pragma unroll
  for (int mi = 0; mi < 2; ++mi) {
#pragma unroll
    for (int ni = 0; ni < 4; ++ni) {
      const int col = n0 + wn * 64 + ni * 16 + fr;
      const float bv = bias ? bias[col] : 0.f;
#pragma unroll
      for (int r = 0; r < 8; ++r) {
        const int row = m0 + wm * 32 + mi * 16 + fh * 8 + r;
        float v = (acc[mi][ni][r] + bv) * scale;
        if (act == 1) v = 0.5f * v * (1.f + erff(v * 0.70710678118654752f));
        if (outF) outF[(size_t)row * N + col] = v;
        if (outB) outB[(size_t)row * N + col] = (bf16_t)v;
      }
    }
  }
}

// ---------------------------------------------------------------------------
// Sliding-window attention (flash-style, WMMA).
// grid = (NB, H, B); 256 threads = 8 waves; wave handles 32 query rows.
// 24 window key-tiles of 32 + 1 final "global token" tile.
// K tile copied with async global->LDS; V transposed through VGPRs.
// ---------------------------------------------------------------------------
__global__ __launch_bounds__(256)
void swattn_kernel(const bf16_t* __restrict__ q,   // [B,S,D] (already / 8)
                   const bf16_t* __restrict__ k,   // [B,S,D]
                   const bf16_t* __restrict__ v,   // [B,S,D]
                   const int*    __restrict__ amask,
                   bf16_t* __restrict__ ctx)       // [B,S,D]
{
  const int nb    = blockIdx.x;
  const int head  = blockIdx.y;
  const int batch = blockIdx.z;
  const int t     = threadIdx.x;
  const int lane  = t & 31;
  const int wave  = t >> 5;
  const int fr    = lane & 15;
  const int fh    = lane >> 4;

  __shared__ bf16_t Ks[32 * 72];      // [key][d]   stride 72 (144B)
  __shared__ bf16_t Vt[64 * 40];      // [d][key]   stride 40 (80B)
  __shared__ bf16_t Ps[8][32 * 40];   // per-wave P [row][key]

  // Q fragments: rows (wave*32 + mi*16 + fr), K-dim = DH split into 2x32
  Frag qa[2][2];
  const size_t qbase = ((size_t)batch * Ss + (size_t)nb * Cc) * Dd + head * DHh;
#pragma unroll
  for (int mi = 0; mi < 2; ++mi)
#pragma unroll
    for (int ks = 0; ks < 2; ++ks) {
      const bf16_t* p = q + qbase + (size_t)(wave * 32 + mi * 16 + fr) * Dd + ks * 32;
      qa[mi][ks].u[0] = *(const uint4*)(p + fh * 8);
      qa[mi][ks].u[1] = *(const uint4*)(p + 16 + fh * 8);
    }

  v8f cacc[2][4];
  float mrow[2][8], lrow[2][8];
#pragma unroll
  for (int mi = 0; mi < 2; ++mi) {
#pragma unroll
    for (int ni = 0; ni < 4; ++ni) cacc[mi][ni] = vzero8();
#pragma unroll
    for (int r = 0; r < 8; ++r) { mrow[mi][r] = -1e30f; lrow[mi][r] = 0.f; }
  }
  const int gmaskv = amask[(size_t)batch * Ss] > 0;

  for (int it = 0; it < 25; ++it) {
    const int  j0 = it * 32;
    const bool gl = (it == 24);       // global-token pseudo tile

    __syncthreads();
    {   // cooperative K/V tile load: K async to LDS, V transposed via VGPRs
      const int kr  = t >> 3;         // 0..31 key row
      const int seg = (t & 7) * 8;    // d offset
      int skey = gl ? (kr == 0 ? 0 : -1) : (nb - 1) * Cc + j0 + kr;
      int sc = skey < 0 ? 0 : (skey > Ss - 1 ? Ss - 1 : skey);
      const size_t gro = ((size_t)batch * Ss + sc) * Dd + head * DHh + seg;
      async_copy_b128(lds_off(Ks + kr * 72 + seg), (uint64_t)(uintptr_t)(k + gro));
      union { uint4 u; bf16_t e[8]; } vu;
      vu.u = *(const uint4*)(v + gro);
#pragma unroll
      for (int e = 0; e < 8; ++e) Vt[(seg + e) * 40 + kr] = vu.e[e];
      wait_async_0();                 // own K slice landed; barrier makes all visible
    }
    __syncthreads();

    // ---- scores: S(32x32) = Q(32x64) @ K_tile(32x64)^T ----
    Frag kb[2][2];
#pragma unroll
    for (int ni = 0; ni < 2; ++ni)
#pragma unroll
      for (int ks = 0; ks < 2; ++ks) {
        const bf16_t* p = Ks + (ni * 16 + fr) * 72 + ks * 32;
        kb[ni][ks].u[0] = *(const uint4*)(p + fh * 8);
        kb[ni][ks].u[1] = *(const uint4*)(p + 16 + fh * 8);
      }
    v8f sacc[2][2];
#pragma unroll
    for (int mi = 0; mi < 2; ++mi)
#pragma unroll
      for (int ni = 0; ni < 2; ++ni) {
        sacc[mi][ni] = vzero8();
#pragma unroll
        for (int ks = 0; ks < 2; ++ks)
          sacc[mi][ni] = wmma_bf16(qa[mi][ks].v, kb[ni][ks].v, sacc[mi][ni]);
      }

    // ---- band/mask + online softmax; write P tile to LDS ----
    bool keyok[2]; int jn[2];
#pragma unroll
    for (int ni = 0; ni < 2; ++ni) {
      jn[ni] = j0 + ni * 16 + fr;
      if (gl) {
        keyok[ni] = (ni == 0) && (fr == 0) && gmaskv;
      } else {
        int skey = (nb - 1) * Cc + jn[ni];
        keyok[ni] = (skey >= 1) && (skey < Ss) &&
                    (amask[(size_t)batch * Ss + skey] > 0);
      }
    }
#pragma unroll
    for (int mi = 0; mi < 2; ++mi) {
#pragma unroll
      for (int r = 0; r < 8; ++r) {
        const int row = wave * 32 + mi * 16 + fh * 8 + r;   // cc in [0,256)
        float s0 = (keyok[0] && (gl || (jn[0] >= row && jn[0] <= row + 2 * Cc)))
                       ? sacc[mi][0][r] : -1e30f;
        float s1 = (keyok[1] && (gl || (jn[1] >= row && jn[1] <= row + 2 * Cc)))
                       ? sacc[mi][1][r] : -1e30f;
        float mx = fmaxf(s0, s1);
#pragma unroll
        for (int off = 1; off < 16; off <<= 1) mx = fmaxf(mx, __shfl_xor(mx, off, 32));
        const float nm   = fmaxf(mrow[mi][r], mx);
        const float corr = expf(mrow[mi][r] - nm);
        lrow[mi][r] *= corr;
#pragma unroll
        for (int ni = 0; ni < 4; ++ni) cacc[mi][ni][r] *= corr;
        float p0 = (s0 > -1e29f) ? expf(s0 - nm) : 0.f;
        float p1 = (s1 > -1e29f) ? expf(s1 - nm) : 0.f;
        float ps = p0 + p1;
#pragma unroll
        for (int off = 1; off < 16; off <<= 1) ps += __shfl_xor(ps, off, 32);
        lrow[mi][r] += ps;
        mrow[mi][r]  = nm;
        bf16_t* prow = &Ps[wave][(mi * 16 + fh * 8 + r) * 40];
        prow[fr]      = (bf16_t)p0;
        prow[16 + fr] = (bf16_t)p1;
      }
    }

    // ---- ctx += P(32x32) @ V_tile(32x64)  (P reloaded in A-layout) ----
    Frag pa[2], vb[4];
#pragma unroll
    for (int mi = 0; mi < 2; ++mi) {
      const bf16_t* p = &Ps[wave][(mi * 16 + fr) * 40];
      pa[mi].u[0] = *(const uint4*)(p + fh * 8);
      pa[mi].u[1] = *(const uint4*)(p + 16 + fh * 8);
    }
#pragma unroll
    for (int ni = 0; ni < 4; ++ni) {
      const bf16_t* p = Vt + (ni * 16 + fr) * 40;
      vb[ni].u[0] = *(const uint4*)(p + fh * 8);
      vb[ni].u[1] = *(const uint4*)(p + 16 + fh * 8);
    }
#pragma unroll
    for (int mi = 0; mi < 2; ++mi)
#pragma unroll
      for (int ni = 0; ni < 4; ++ni)
        cacc[mi][ni] = wmma_bf16(pa[mi].v, vb[ni].v, cacc[mi][ni]);
  }

  // ---- finalize: ctx /= l ----
#pragma unroll
  for (int mi = 0; mi < 2; ++mi)
#pragma unroll
    for (int r = 0; r < 8; ++r) {
      const float inv = lrow[mi][r] > 0.f ? 1.f / lrow[mi][r] : 0.f;
      const int row = nb * Cc + wave * 32 + mi * 16 + fh * 8 + r;
#pragma unroll
      for (int ni = 0; ni < 4; ++ni)
        ctx[((size_t)batch * Ss + row) * Dd + head * DHh + ni * 16 + fr] =
            (bf16_t)(cacc[mi][ni][r] * inv);
    }
}

// ---------------------------------------------------------------------------
// Full attention for query position 0 (overwrites ctx[b, 0, head])
// grid = (H, B), 256 threads
// ---------------------------------------------------------------------------
__global__ __launch_bounds__(256)
void row0_attn_kernel(const bf16_t* __restrict__ q, const bf16_t* __restrict__ k,
                      const bf16_t* __restrict__ v, const int* __restrict__ amask,
                      bf16_t* __restrict__ ctx) {
  const int head = blockIdx.x, batch = blockIdx.y;
  const int t = threadIdx.x;
  __shared__ float q0[64];
  __shared__ float red[256];
  __shared__ float lred[4];
  const size_t base = (size_t)batch * Ss * Dd + head * DHh;
  if (t < 64) q0[t] = (float)q[base + t];
  __syncthreads();

  // pass 1: global max of masked scores
  float mx = -1e30f;
  for (int s = t; s < Ss; s += 256) {
    if (amask[(size_t)batch * Ss + s] > 0) {
      const bf16_t* kp = k + base + (size_t)s * Dd;
      float d = 0.f;
#pragma unroll
      for (int e = 0; e < 64; ++e) d += q0[e] * (float)kp[e];
      mx = fmaxf(mx, d);
    }
  }
  red[t] = mx; __syncthreads();
  for (int o = 128; o > 0; o >>= 1) { if (t < o) red[t] = fmaxf(red[t], red[t + o]); __syncthreads(); }
  const float gmax = red[0];
  __syncthreads();

  // pass 2: thread (g,d) accumulates p*v over strided keys
  const int g = t >> 6, d = t & 63;
  float acc = 0.f, lsum = 0.f;
  for (int s = g; s < Ss; s += 4) {
    if (amask[(size_t)batch * Ss + s] > 0) {
      const bf16_t* kp = k + base + (size_t)s * Dd;
      float dt = 0.f;
#pragma unroll
      for (int e = 0; e < 64; ++e) dt += q0[e] * (float)kp[e];
      const float p = expf(dt - gmax);
      lsum += p;
      acc  += p * (float)v[base + (size_t)s * Dd + d];
    }
  }
  red[t] = acc;
  if (d == 0) lred[g] = lsum;
  __syncthreads();
  if (t < 64) {
    const float a = red[t] + red[64 + t] + red[128 + t] + red[192 + t];
    const float L = lred[0] + lred[1] + lred[2] + lred[3];
    ctx[base + t] = (bf16_t)(L > 0.f ? a / L : 0.f);
  }
}

// ---------------------------------------------------------------------------
// logits[b,j] = h[b, 0, :] . cls_W[:, j] + cls_b[j]
// ---------------------------------------------------------------------------
__global__ void cls_kernel(const float* __restrict__ h, const float* __restrict__ W,
                           const float* __restrict__ bvec, float* __restrict__ out) {
  const int t = threadIdx.x;
  if (t < Bb * 3) {
    const int b = t / 3, j = t % 3;
    const float* hr = h + (size_t)b * Ss * Dd;   // position 0 row
    float s = bvec[j];
    for (int d2 = 0; d2 < Dd; ++d2) s += hr[d2] * W[d2 * 3 + j];
    out[t] = s;
  }
}

// ---------------------------------------------------------------------------
// Host launcher
// ---------------------------------------------------------------------------
extern "C" void kernel_launch(void* const* d_in, const int* in_sizes, int n_in,
                              void* d_out, int out_size, void* d_ws, size_t ws_size,
                              hipStream_t stream) {
  (void)in_sizes; (void)n_in; (void)out_size; (void)ws_size;

  const int*   input_ids = (const int*)d_in[0];
  const int*   amask     = (const int*)d_in[1];
  const float* emb_tok   = (const float*)d_in[2];
  const float* emb_pos   = (const float*)d_in[3];
  const float* emb_g     = (const float*)d_in[4];
  const float* emb_b     = (const float*)d_in[5];
  const float* Wq        = (const float*)d_in[6];
  const float* bq        = (const float*)d_in[7];
  const float* Wk        = (const float*)d_in[8];
  const float* bk        = (const float*)d_in[9];
  const float* Wv        = (const float*)d_in[10];
  const float* bv        = (const float*)d_in[11];
  const float* Wo        = (const float*)d_in[12];
  const float* bo        = (const float*)d_in[13];
  const float* ln1_g     = (const float*)d_in[14];
  const float* ln1_b     = (const float*)d_in[15];
  const float* W1        = (const float*)d_in[16];
  const float* b1        = (const float*)d_in[17];
  const float* W2        = (const float*)d_in[18];
  const float* b2        = (const float*)d_in[19];
  const float* ln2_g     = (const float*)d_in[20];
  const float* ln2_b     = (const float*)d_in[21];
  const float* cls_W     = (const float*)d_in[22];
  const float* cls_b     = (const float*)d_in[23];

  uint8_t* wsp = (uint8_t*)d_ws;
  auto take = [&](size_t bytes) -> void* {
    void* p = (void*)wsp;
    wsp += (bytes + 255) & ~(size_t)255;
    return p;
  };
  float*  hF   = (float*) take((size_t)Mm * Dd * 4);
  bf16_t* hB   = (bf16_t*)take((size_t)Mm * Dd * 2);
  bf16_t* qB   = (bf16_t*)take((size_t)Mm * Dd * 2);
  bf16_t* kB   = (bf16_t*)take((size_t)Mm * Dd * 2);
  bf16_t* vB   = (bf16_t*)take((size_t)Mm * Dd * 2);
  bf16_t* ctxB = (bf16_t*)take((size_t)Mm * Dd * 2);
  float*  aF   = (float*) take((size_t)Mm * Dd * 4);
  bf16_t* f1B  = (bf16_t*)take((size_t)Mm * Ff * 2);
  bf16_t* WqT  = (bf16_t*)take((size_t)Ll * Dd * Dd * 2);
  bf16_t* WkT  = (bf16_t*)take((size_t)Ll * Dd * Dd * 2);
  bf16_t* WvT  = (bf16_t*)take((size_t)Ll * Dd * Dd * 2);
  bf16_t* WoT  = (bf16_t*)take((size_t)Ll * Dd * Dd * 2);
  bf16_t* W1T  = (bf16_t*)take((size_t)Ll * Dd * Ff * 2);
  bf16_t* W2T  = (bf16_t*)take((size_t)Ll * Ff * Dd * 2);

  const int nDD = Dd * Dd, nDF = Dd * Ff;
  const int gDD = (nDD + 255) / 256, gDF = (nDF + 255) / 256;
  for (int l = 0; l < Ll; ++l) {
    transpose_bf16_kernel<<<gDD, 256, 0, stream>>>(Wq + (size_t)l * nDD, WqT + (size_t)l * nDD, Dd, Dd);
    transpose_bf16_kernel<<<gDD, 256, 0, stream>>>(Wk + (size_t)l * nDD, WkT + (size_t)l * nDD, Dd, Dd);
    transpose_bf16_kernel<<<gDD, 256, 0, stream>>>(Wv + (size_t)l * nDD, WvT + (size_t)l * nDD, Dd, Dd);
    transpose_bf16_kernel<<<gDD, 256, 0, stream>>>(Wo + (size_t)l * nDD, WoT + (size_t)l * nDD, Dd, Dd);
    transpose_bf16_kernel<<<gDF, 256, 0, stream>>>(W1 + (size_t)l * nDF, W1T + (size_t)l * nDF, Dd, Ff);
    transpose_bf16_kernel<<<gDF, 256, 0, stream>>>(W2 + (size_t)l * nDF, W2T + (size_t)l * nDF, Ff, Dd);
  }

  embed_ln_kernel<<<Mm, 256, 0, stream>>>(input_ids, emb_tok, emb_pos, emb_g, emb_b, hF, hB);

  const dim3 gProj(Mm / 128, Dd / 128);   // 64 x 6
  const dim3 gFF1 (Mm / 128, Ff / 128);   // 64 x 24
  for (int l = 0; l < Ll; ++l) {
    const bf16_t* wq = WqT + (size_t)l * nDD;
    const bf16_t* wk = WkT + (size_t)l * nDD;
    const bf16_t* wv = WvT + (size_t)l * nDD;
    const bf16_t* wo = WoT + (size_t)l * nDD;
    const bf16_t* w1 = W1T + (size_t)l * nDF;
    const bf16_t* w2 = W2T + (size_t)l * nDF;

    gemm_bf16_kernel<<<gProj, 256, 0, stream>>>(hB, wq, bq + l * Dd, nullptr, qB,
                                                Mm, Dd, Dd, 0.125f, 0);   // Q / sqrt(DH)
    gemm_bf16_kernel<<<gProj, 256, 0, stream>>>(hB, wk, bk + l * Dd, nullptr, kB,
                                                Mm, Dd, Dd, 1.f, 0);
    gemm_bf16_kernel<<<gProj, 256, 0, stream>>>(hB, wv, bv + l * Dd, nullptr, vB,
                                                Mm, Dd, Dd, 1.f, 0);

    swattn_kernel<<<dim3(NBb, Hh, Bb), 256, 0, stream>>>(qB, kB, vB, amask, ctxB);
    row0_attn_kernel<<<dim3(Hh, Bb), 256, 0, stream>>>(qB, kB, vB, amask, ctxB);

    gemm_bf16_kernel<<<gProj, 256, 0, stream>>>(ctxB, wo, bo + l * Dd, aF, nullptr,
                                                Mm, Dd, Dd, 1.f, 0);
    add_ln_kernel<<<Mm, 256, 0, stream>>>(hF, aF, ln1_g + l * Dd, ln1_b + l * Dd, hF, hB);

    gemm_bf16_kernel<<<gFF1, 256, 0, stream>>>(hB, w1, b1 + l * Ff, nullptr, f1B,
                                               Mm, Ff, Dd, 1.f, 1);       // GELU
    gemm_bf16_kernel<<<gProj, 256, 0, stream>>>(f1B, w2, b2 + l * Dd, aF, nullptr,
                                                Mm, Dd, Ff, 1.f, 0);
    add_ln_kernel<<<Mm, 256, 0, stream>>>(hF, aF, ln2_g + l * Dd, ln2_b + l * Dd, hF, hB);
  }

  cls_kernel<<<1, 64, 0, stream>>>(hF, cls_W, cls_b, (float*)d_out);
}
